// SRNN_9414568313028
// MI455X (gfx1250) — matrix operand order repaired and checked by reading, not compiled, not run on hardware
//
#include <hip/hip_runtime.h>
#include <hip/hip_bf16.h>
#include <stdint.h>

// ---------------------------------------------------------------------------
// SRNN (social-LSTM style) for MI455X / gfx1250.
// bf16 WMMA GEMMs (v_wmma_f32_16x16x32_bf16), fp32 accumulate + elementwise.
// Weights pre-swizzled into WMMA B-fragment order; staged to LDS with
// global_load_async_to_lds_b128 (ASYNCcnt) per CDNA5 ISA.
// Node kernel: 4 waves/block (one per SIMD32 of the WGP), 304 KB LDS.
// ---------------------------------------------------------------------------

typedef __attribute__((ext_vector_type(16))) __bf16 bf16x16;
typedef __attribute__((ext_vector_type(8)))  float  f32x8;

#define S_LEN   16
#define N_NODES 32768
#define E_EDGES 131072

static __device__ __forceinline__ float sigmoidf_(float x) {
    return 1.0f / (1.0f + __expf(-x));
}
static __device__ __forceinline__ float tanhf_(float x) {
    float e = __expf(-2.0f * x);
    return (1.0f - e) / (1.0f + e);
}

// Low 32 bits of a flat pointer into LDS == LDS byte offset (aperture layout).
static __device__ __forceinline__ uint32_t lds_off(const void* p) {
    return (uint32_t)(uintptr_t)p;
}

// CDNA5 async copy global -> LDS, 16 bytes per active lane. Tracked by ASYNCcnt.
static __device__ __forceinline__ void async_copy16(uint32_t lds_addr, const void* gaddr) {
    asm volatile("global_load_async_to_lds_b128 %0, %1, off"
                 :: "v"(lds_addr), "v"(gaddr) : "memory");
}
static __device__ __forceinline__ void async_wait0() {
    asm volatile("s_wait_asynccnt 0" ::: "memory");
}

// A-fragment K mapping for 16-bit A (16x32), wave32 (ISA 7.12.2):
// element j (= 2v+h), lanes 0-15: K = {0..7,16..23}; lanes 16-31: +8.
static __device__ __forceinline__ int a_kmap(int j, int laneHi) {
    int v = j >> 1, h = j & 1;
    int base = (v < 4) ? (2 * v) : (16 + 2 * (v - 4));
    return base + h + (laneHi ? 8 : 0);
}

// ---------------------------------------------------------------------------
// Weight prep: convert fp32 [w0; w1] (stacked along K) to bf16, laid out in
// WMMA B-fragment order. Chunk = (kc * numTiles + t), 512 halves per chunk;
// within a chunk lane L holds 16 consecutive halves:
//   K = kc*32 + (L>=16 ? 8 : 0) + (j<8 ? j : j+8),  N = t*16 + (L&15).
// ---------------------------------------------------------------------------
__global__ void prep_swz_kernel(uint16_t* __restrict__ dst_u16,
                                const float* __restrict__ w0,
                                const float* __restrict__ w1,
                                int K1, int Ktot, int Ncols) {
    __bf16* dst = (__bf16*)dst_u16;
    int idx = blockIdx.x * blockDim.x + threadIdx.x;
    int total = Ktot * Ncols;
    if (idx >= total) return;
    int numTiles = Ncols >> 4;
    int chunk = idx >> 9;
    int off   = idx & 511;
    int kc = chunk / numTiles;
    int t  = chunk - kc * numTiles;
    int L  = off >> 4;
    int j  = off & 15;
    int K  = kc * 32 + ((L >= 16) ? 8 : 0) + ((j < 8) ? j : j + 8);
    int n  = t * 16 + (L & 15);
    float v = (K < K1) ? w0[K * Ncols + n] : w1[(K - K1) * Ncols + n];
    dst[idx] = (__bf16)v;
}

__global__ void zero_kernel(float4* __restrict__ p, int n4) {
    int i = blockIdx.x * blockDim.x + threadIdx.x;
    if (i < n4) p[i] = make_float4(0.f, 0.f, 0.f, 0.f);
}

// segment_sum: agg[node] += hs[edge]   (2E incidences, edge = m>>1)
__global__ void scatter_kernel(const float* __restrict__ hs,
                               const int* __restrict__ inc,
                               float* __restrict__ agg) {
    int gid = blockIdx.x * blockDim.x + threadIdx.x;   // 2E*16 threads
    int m   = gid >> 4;
    int c4  = (gid & 15) << 2;
    int node = inc[m];
    int edge = m >> 1;
    const float4 v = *(const float4*)(hs + (size_t)edge * 64 + c4);
    float* dstp = agg + (size_t)node * 64 + c4;
    atomicAdd(dstp + 0, v.x);
    atomicAdd(dstp + 1, v.y);
    atomicAdd(dstp + 2, v.z);
    atomicAdd(dstp + 3, v.w);
}

// ---------------------------------------------------------------------------
// Edge LSTM (temporal & spatial). Hidden 64, gates 256, input enc(64)+h(64).
// A = [relu(x@enc_w+b) | h]  (16x128 bf16),  B = [Wih;Whh] (128x256 bf16).
// 256 threads = 8 waves, each wave owns one 16-row tile (128 rows/block).
// ---------------------------------------------------------------------------
#define EDGE_WAVES 8
__global__ __launch_bounds__(256) void edge_lstm_kernel(
    const float* __restrict__ x2, const float* __restrict__ enc_w,
    const float* __restrict__ enc_b, const uint16_t* __restrict__ Bswz_u16,
    const float* __restrict__ bias, float* __restrict__ h, float* __restrict__ c) {
    extern __shared__ __align__(128) char smem[];
    __bf16* lB = (__bf16*)smem;                   // 32768 halves (64 KB)
    __bf16* lA = (__bf16*)(smem + 65536);         // 8 waves * 2048 halves

    const int tid = threadIdx.x;
    {   // stage B into LDS with async copies (16 B/lane/op)
        const char* src = (const char*)Bswz_u16;
        uint32_t dstBase = lds_off(lB);
        for (int i = tid; i < 4096; i += 256)
            async_copy16(dstBase + i * 16, src + i * 16);
        async_wait0();
    }
    __syncthreads();

    const int w  = tid >> 5;
    const int L  = tid & 31;
    const int Ln = L & 15;
    const int hi = L >> 4;
    const int r0 = (blockIdx.x * EDGE_WAVES + w) * 16;
    __bf16* A = lA + w * 2048;

    // enc -> A[:,0:64], h -> A[:,64:128]
    for (int p = 0; p < 32; ++p) {
        int idx = L + p * 32;                 // 0..1023
        int row = idx >> 6, col = idx & 63;
        float x0 = x2[(size_t)(r0 + row) * 2 + 0];
        float x1 = x2[(size_t)(r0 + row) * 2 + 1];
        float e  = x0 * enc_w[col] + x1 * enc_w[64 + col] + enc_b[col];
        A[row * 128 + col]       = (__bf16)fmaxf(e, 0.0f);
        A[row * 128 + 64 + col]  = (__bf16)h[(size_t)(r0 + row) * 64 + col];
    }
    // LDS ops are in-order within a wave; safe to gather A fragments now.
    union { bf16x16 v; __bf16 e[16]; } aF[4];
    for (int kc = 0; kc < 4; ++kc)
        for (int j = 0; j < 16; ++j)
            aF[kc].e[j] = A[Ln * 128 + kc * 32 + a_kmap(j, hi)];

    const f32x8 zero8 = {0.f,0.f,0.f,0.f,0.f,0.f,0.f,0.f};
    for (int jb = 0; jb < 4; ++jb) {
        f32x8 acc[4] = {zero8, zero8, zero8, zero8};
        for (int kc = 0; kc < 4; ++kc) {
#pragma unroll
            for (int g = 0; g < 4; ++g) {
                int t = g * 4 + jb;
                const bf16x16 bF = *(const bf16x16*)(lB + (kc * 16 + t) * 512 + L * 16);
                acc[g] = __builtin_amdgcn_wmma_f32_16x16x32_bf16(
                    false, aF[kc].v, false, bF, (short)0, acc[g], false, false);
            }
        }
        int col = jb * 16 + Ln;
        float bi = bias[col], bf = bias[64 + col], bg = bias[128 + col], bo = bias[192 + col];
#pragma unroll
        for (int r = 0; r < 8; ++r) {
            int gr = r0 + r + hi * 8;
            float iv = acc[0][r] + bi;
            float fv = acc[1][r] + bf;
            float gv = acc[2][r] + bg;
            float ov = acc[3][r] + bo;
            float cold = c[(size_t)gr * 64 + col];
            float c2 = sigmoidf_(fv) * cold + sigmoidf_(iv) * tanhf_(gv);
            float h2 = sigmoidf_(ov) * tanhf_(c2);
            c[(size_t)gr * 64 + col] = c2;
            h[(size_t)gr * 64 + col] = h2;
        }
    }
}

// ---------------------------------------------------------------------------
// Node kernel: fuses ee-GEMM (emb = relu([ht|agg]@ee_w+b)), node encoder,
// node LSTM (hidden 128, K=256 combined [enc|emb|hn], gates 512), and the
// output projection out = hn@out_w + out_b (16-lane shfl reduction).
// 128 threads = 4 waves (one per SIMD32). Per-wave 8 KB staging buffer is
// reused: first as [ht|agg] (16x128), then (after aT fragments are in
// registers) as A = [enc|emb|hn] (16x256). LDS total: 304 KB.
// ---------------------------------------------------------------------------
#define NODE_WAVES 4
__global__ __launch_bounds__(128) void node_lstm_kernel(
    const float* __restrict__ xn, const float* __restrict__ n_enc_w,
    const float* __restrict__ n_enc_b,
    const float* __restrict__ ht, const float* __restrict__ agg,
    const uint16_t* __restrict__ eBswz, const float* __restrict__ ee_b,
    const uint16_t* __restrict__ nBswz, const float* __restrict__ n_b,
    float* __restrict__ hn, float* __restrict__ cn,
    const float* __restrict__ out_w, const float* __restrict__ out_b,
    float* __restrict__ out, int numWavesTotal) {
    extern __shared__ __align__(128) char smem[];
    __bf16* nB = (__bf16*)smem;                          // 131072 halves (256 KB)
    __bf16* eB = (__bf16*)(smem + 262144);               // 8192 halves (16 KB)
    __bf16* lA = (__bf16*)(smem + 262144 + 16384);       // 4 waves * 4096 halves

    const int tid = threadIdx.x;
    {
        uint32_t d0 = lds_off(nB);
        const char* s0 = (const char*)nBswz;
        for (int i = tid; i < 16384; i += 128) async_copy16(d0 + i * 16, s0 + i * 16);
        uint32_t d1 = lds_off(eB);
        const char* s1 = (const char*)eBswz;
        for (int i = tid; i < 1024; i += 128) async_copy16(d1 + i * 16, s1 + i * 16);
        async_wait0();
    }
    __syncthreads();

    const int w  = tid >> 5;
    const int L  = tid & 31;
    const int Ln = L & 15;
    const int hi = L >> 4;
    __bf16* A = lA + w * 4096;     // 16x256; first 2048 halves double as T (16x128)
    const int waveId = blockIdx.x * NODE_WAVES + w;
    const f32x8 zero8 = {0.f,0.f,0.f,0.f,0.f,0.f,0.f,0.f};

    for (int tile = waveId; tile < (N_NODES / 16); tile += numWavesTotal) {
        const int r0 = tile * 16;
        // stage T = [ht | agg] (16 x 128) as bf16 into first half of A buffer
        for (int p = 0; p < 64; ++p) {
            int idx = L + p * 32;            // 0..2047
            int row = idx >> 7, col = idx & 127;
            float v = (col < 64) ? ht[(size_t)(r0 + row) * 64 + col]
                                 : agg[(size_t)(r0 + row) * 64 + (col - 64)];
            A[row * 128 + col] = (__bf16)v;
        }
        union { bf16x16 v; __bf16 e[16]; } aT[4];
        for (int kc = 0; kc < 4; ++kc)
            for (int j = 0; j < 16; ++j)
                aT[kc].e[j] = A[Ln * 128 + kc * 32 + a_kmap(j, hi)];
        // aT now in registers; A buffer free for reuse (wave-local LDS in-order).

        // emb = relu([ht|agg] @ ee_w + ee_b) -> A cols 64..127
        for (int t4 = 0; t4 < 4; ++t4) {
            f32x8 acc = zero8;
            for (int kc = 0; kc < 4; ++kc) {
                const bf16x16 bF = *(const bf16x16*)(eB + (kc * 4 + t4) * 512 + L * 16);
                acc = __builtin_amdgcn_wmma_f32_16x16x32_bf16(
                    false, aT[kc].v, false, bF, (short)0, acc, false, false);
            }
            int col = t4 * 16 + Ln;
#pragma unroll
            for (int r = 0; r < 8; ++r) {
                int row = r + hi * 8;
                float v = acc[r] + ee_b[col];
                A[row * 256 + 64 + col] = (__bf16)fmaxf(v, 0.f);
            }
        }
        // enc = relu(xn * n_enc_w + b) -> A cols 0..63
        for (int p = 0; p < 32; ++p) {
            int idx = L + p * 32;            // 0..1023
            int row = idx >> 6, col = idx & 63;
            float v = xn[r0 + row] * n_enc_w[col] + n_enc_b[col];
            A[row * 256 + col] = (__bf16)fmaxf(v, 0.f);
        }
        // hn -> A cols 128..255
        for (int p = 0; p < 64; ++p) {
            int idx = L + p * 32;            // 0..2047
            int row = idx >> 7, col = idx & 127;
            A[row * 256 + 128 + col] = (__bf16)hn[(size_t)(r0 + row) * 128 + col];
        }
        union { bf16x16 v; __bf16 e[16]; } aN[8];
        for (int kc = 0; kc < 8; ++kc)
            for (int j = 0; j < 16; ++j)
                aN[kc].e[j] = A[Ln * 256 + kc * 32 + a_kmap(j, hi)];

        float po[8] = {0.f,0.f,0.f,0.f,0.f,0.f,0.f,0.f};
        for (int jb = 0; jb < 8; ++jb) {
            f32x8 acc[4] = {zero8, zero8, zero8, zero8};
            for (int kc = 0; kc < 8; ++kc) {
#pragma unroll
                for (int g = 0; g < 4; ++g) {
                    int t = g * 8 + jb;
                    const bf16x16 bF = *(const bf16x16*)(nB + (kc * 32 + t) * 512 + L * 16);
                    acc[g] = __builtin_amdgcn_wmma_f32_16x16x32_bf16(
                        false, aN[kc].v, false, bF, (short)0, acc[g], false, false);
                }
            }
            int col = jb * 16 + Ln;
            float bi = n_b[col], bf = n_b[128 + col], bg = n_b[256 + col], bo = n_b[384 + col];
            float ow = out_w[col];
#pragma unroll
            for (int r = 0; r < 8; ++r) {
                int gr = r0 + r + hi * 8;
                float iv = acc[0][r] + bi;
                float fv = acc[1][r] + bf;
                float gv = acc[2][r] + bg;
                float ov = acc[3][r] + bo;
                float cold = cn[(size_t)gr * 128 + col];
                float c2 = sigmoidf_(fv) * cold + sigmoidf_(iv) * tanhf_(gv);
                float h2 = sigmoidf_(ov) * tanhf_(c2);
                cn[(size_t)gr * 128 + col] = c2;
                hn[(size_t)gr * 128 + col] = h2;
                po[r] += h2 * ow;
            }
        }
        float ob = out_b[0];
#pragma unroll
        for (int r = 0; r < 8; ++r) {
            float s = po[r];
            s += __shfl_xor(s, 1, 16);
            s += __shfl_xor(s, 2, 16);
            s += __shfl_xor(s, 4, 16);
            s += __shfl_xor(s, 8, 16);
            if (Ln == 0) out[r0 + r + hi * 8] = s + ob;
        }
    }
}

// ---------------------------------------------------------------------------
// Host launch
// ---------------------------------------------------------------------------
extern "C" void kernel_launch(void* const* d_in, const int* in_sizes, int n_in,
                              void* d_out, int out_size, void* d_ws, size_t ws_size,
                              hipStream_t stream) {
    (void)in_sizes; (void)n_in; (void)out_size; (void)ws_size;
    const float* data_nodes = (const float*)d_in[0];
    const float* data_t     = (const float*)d_in[1];
    const float* data_s     = (const float*)d_in[2];
    const float* h_n0 = (const float*)d_in[3];
    const float* c_n0 = (const float*)d_in[4];
    const float* h_t0 = (const float*)d_in[5];
    const float* c_t0 = (const float*)d_in[6];
    const float* h_s0 = (const float*)d_in[7];
    const float* c_s0 = (const float*)d_in[8];
    const int*   inc  = (const int*)d_in[9];
    const float* t_enc_w = (const float*)d_in[10];
    const float* t_enc_b = (const float*)d_in[11];
    const float* t_Wih   = (const float*)d_in[12];
    const float* t_Whh   = (const float*)d_in[13];
    const float* t_b     = (const float*)d_in[14];
    const float* s_enc_w = (const float*)d_in[15];
    const float* s_enc_b = (const float*)d_in[16];
    const float* s_Wih   = (const float*)d_in[17];
    const float* s_Whh   = (const float*)d_in[18];
    const float* s_b     = (const float*)d_in[19];
    const float* n_enc_w = (const float*)d_in[20];
    const float* n_enc_b = (const float*)d_in[21];
    const float* ee_w    = (const float*)d_in[22];
    const float* ee_b    = (const float*)d_in[23];
    const float* n_Wih   = (const float*)d_in[24];
    const float* n_Whh   = (const float*)d_in[25];
    const float* n_b     = (const float*)d_in[26];
    const float* out_w   = (const float*)d_in[27];
    const float* out_b   = (const float*)d_in[28];
    float* out = (float*)d_out;

    char* ws = (char*)d_ws;
    uint16_t* W_T = (uint16_t*)(ws + 0);        //  64 KB  (128x256 bf16 swizzled)
    uint16_t* W_S = (uint16_t*)(ws + 65536);    //  64 KB
    uint16_t* W_N = (uint16_t*)(ws + 131072);   // 256 KB  (256x512)
    uint16_t* W_E = (uint16_t*)(ws + 393216);   //  16 KB  (128x64)
    float* AGG = (float*)(ws + 409600);                       // N*64  (8 MB)
    float* HT  = (float*)(ws + 409600ull + 8388608ull);       // N*64
    float* CT  = (float*)(ws + 409600ull + 2ull * 8388608ull);
    float* HS  = (float*)(ws + 409600ull + 3ull * 8388608ull);             // E*64 (32 MB)
    float* CS  = (float*)(ws + 409600ull + 3ull * 8388608ull + 33554432ull);
    float* HN  = (float*)(ws + 409600ull + 3ull * 8388608ull + 2ull * 33554432ull); // N*128
    float* CN  = (float*)(ws + 409600ull + 3ull * 8388608ull + 2ull * 33554432ull + 16777216ull);

    // carry state init (inputs must not be mutated)
    hipMemcpyAsync(HT, h_t0, (size_t)N_NODES * 64 * 4, hipMemcpyDeviceToDevice, stream);
    hipMemcpyAsync(CT, c_t0, (size_t)N_NODES * 64 * 4, hipMemcpyDeviceToDevice, stream);
    hipMemcpyAsync(HS, h_s0, (size_t)E_EDGES * 64 * 4, hipMemcpyDeviceToDevice, stream);
    hipMemcpyAsync(CS, c_s0, (size_t)E_EDGES * 64 * 4, hipMemcpyDeviceToDevice, stream);
    hipMemcpyAsync(HN, h_n0, (size_t)N_NODES * 128 * 4, hipMemcpyDeviceToDevice, stream);
    hipMemcpyAsync(CN, c_n0, (size_t)N_NODES * 128 * 4, hipMemcpyDeviceToDevice, stream);

    // weight prep (bf16 + B-fragment swizzle)
    prep_swz_kernel<<<128, 256, 0, stream>>>(W_T, t_Wih, t_Whh, 64, 128, 256);
    prep_swz_kernel<<<128, 256, 0, stream>>>(W_S, s_Wih, s_Whh, 64, 128, 256);
    prep_swz_kernel<<<512, 256, 0, stream>>>(W_N, n_Wih, n_Whh, 128, 256, 512);
    prep_swz_kernel<<<32,  256, 0, stream>>>(W_E, ee_w, ee_w, 128, 128, 64);

    const size_t edgeSmem = 65536 + EDGE_WAVES * 2048 * 2;                 //  98304 B
    const size_t nodeSmem = 262144 + 16384 + NODE_WAVES * 4096 * 2;        // 311296 B

    for (int step = 0; step < S_LEN; ++step) {
        const float* xt = data_t + (size_t)step * N_NODES * 2;
        const float* xs = data_s + (size_t)step * E_EDGES * 2;
        const float* xn = data_nodes + (size_t)step * N_NODES;
        float* outp = out + (size_t)step * N_NODES;

        zero_kernel<<<(N_NODES * 64 / 4 + 255) / 256, 256, 0, stream>>>(
            (float4*)AGG, N_NODES * 64 / 4);

        edge_lstm_kernel<<<N_NODES / (EDGE_WAVES * 16), 256, edgeSmem, stream>>>(
            xt, t_enc_w, t_enc_b, W_T, t_b, HT, CT);
        edge_lstm_kernel<<<E_EDGES / (EDGE_WAVES * 16), 256, edgeSmem, stream>>>(
            xs, s_enc_w, s_enc_b, W_S, s_b, HS, CS);

        scatter_kernel<<<(2 * E_EDGES * 16) / 256, 256, 0, stream>>>(HS, inc, AGG);

        node_lstm_kernel<<<128, 128, nodeSmem, stream>>>(
            xn, n_enc_w, n_enc_b, HT, AGG, W_E, ee_b, W_N, n_b,
            HN, CN, out_w, out_b, outp, 128 * NODE_WAVES);
    }
}
